// SwinBlock_19791209299983
// MI455X (gfx1250) — compile-verified, matrix operand order checked by reading, not run on hardware
//
#include <hip/hip_runtime.h>
#include <math.h>

// ---------------------------------------------------------------------------
// CDNA5 (gfx1250) SwinV2 forward. bf16 WMMA (v_wmma_f32_16x16x32_bf16) for all
// GEMMs and the fused per-window attention. wave32 everywhere.
// LDS tiles are stored pre-swizzled in WMMA fragment order so every operand
// load is a pair of ds_load_b128. The GEMM is software-pipelined: tile k+1 is
// loaded into registers while tile k runs through the WMMAs.
// ---------------------------------------------------------------------------

typedef __attribute__((ext_vector_type(16))) __bf16 v16bf;
typedef __attribute__((ext_vector_type(8)))  float  v8f;
typedef __attribute__((ext_vector_type(8)))  __bf16 bf16x8;
typedef __attribute__((ext_vector_type(4)))  __bf16 bf16x4;
typedef __attribute__((ext_vector_type(4)))  float  f32x4;

__device__ __forceinline__ v8f wmma_bf16(v16bf a, v16bf b, v8f c) {
  return __builtin_amdgcn_wmma_f32_16x16x32_bf16(false, a, false, b, (short)0, c,
                                                 false, false);
}

// CDNA5 16-bit operand fragment mapping for a 16x32 (A: MxK) / 32x16 (B: KxN)
// tile. Lane = (row|col & 15) + 16*((k>>3)&1); element = (k&7) + 8*(k>>4).
// 4 (and 8) consecutive k are contiguous within a lane's 16-element block.
__device__ __forceinline__ int fragLane(int rc, int k) {
  return (rc & 15) + (((k >> 3) & 1) << 4);
}
__device__ __forceinline__ int fragElem(int k) {
  return (k & 7) + (((k >> 4) & 1) << 3);
}

// ---------------------------------------------------------------------------
// Tiled GEMM: C[M,N] = A[M,K] (f32) * B[K,N] (f32 weights) (+bias) (+gelu)
// bf16 staged in LDS (fragment order), f32 accumulate.
// Block tile 128x64, 8 waves, K-step 32, register-pipelined global loads.
// REQUIRES: M % 128 == 0, K % 32 == 0, N % 4 == 0 (true for every call here).
// ---------------------------------------------------------------------------
__global__ __launch_bounds__(256) void gemm_bf16_kernel(
    const float* __restrict__ A, const float* __restrict__ Bw,
    const float* __restrict__ bias, float* __restrict__ Cout,
    int M, int N, int K, int fuse /*1 = exact GELU*/) {
  __shared__ __bf16 As[8 * 32 * 16];  // [mtile 0..7][lane][elem]
  __shared__ __bf16 Bs[4 * 32 * 16];  // [ntile 0..3][lane][elem]
  int tid  = threadIdx.x;
  int wave = tid >> 5, lane = tid & 31;
  int wm = wave & 3, wn = wave >> 2;
  int laneHi = lane >> 4, lane15 = lane & 15;
  long m0 = (long)blockIdx.x * 128;
  int  n0 = blockIdx.y * 64;
  bool interior = (n0 + 64 <= N);  // uniform: whole B tile in range

  const v8f vz = {0.f, 0.f, 0.f, 0.f, 0.f, 0.f, 0.f, 0.f};
  v8f acc[2][2];
  acc[0][0] = vz; acc[0][1] = vz; acc[1][0] = vz; acc[1][1] = vz;

  f32x4 ra[4], rb[2];
  auto loadTiles = [&](int k0) {
#pragma unroll
    for (int it = 0; it < 4; ++it) {           // A: 128x32, 4 float4/thread
      int idx = tid + it * 256;
      int row = idx >> 3;
      int c4  = (idx & 7) << 2;
      ra[it] = *(const f32x4*)(A + (m0 + row) * K + k0 + c4);
    }
    if (interior) {
#pragma unroll
      for (int it = 0; it < 2; ++it) {         // B: 32x64, 2 float4/thread
        int idx = tid + it * 256;
        int k  = idx >> 4;
        int c4 = (idx & 15) << 2;
        rb[it] = *(const f32x4*)(Bw + (long)(k0 + k) * N + n0 + c4);
      }
    } else {
#pragma unroll
      for (int it = 0; it < 2; ++it) {
        int idx = tid + it * 256;
        int k  = idx >> 4;
        int c4 = (idx & 15) << 2;
        if (n0 + c4 < N) {
          rb[it] = *(const f32x4*)(Bw + (long)(k0 + k) * N + n0 + c4);
        } else {
          rb[it][0] = 0.f; rb[it][1] = 0.f; rb[it][2] = 0.f; rb[it][3] = 0.f;
        }
      }
    }
  };
  auto storeTiles = [&]() {
#pragma unroll
    for (int it = 0; it < 4; ++it) {
      int idx = tid + it * 256;
      int row = idx >> 3;
      int c4  = (idx & 7) << 2;
      bf16x4 h;
      h[0] = (__bf16)ra[it][0]; h[1] = (__bf16)ra[it][1];
      h[2] = (__bf16)ra[it][2]; h[3] = (__bf16)ra[it][3];
      int blk = (row >> 4) * 32 + fragLane(row, c4);
      *(bf16x4*)&As[(blk << 4) + fragElem(c4)] = h;
    }
#pragma unroll
    for (int it = 0; it < 2; ++it) {
      int idx = tid + it * 256;
      int k  = idx >> 4;
      int c4 = (idx & 15) << 2;
      int e  = fragElem(k);
#pragma unroll
      for (int u = 0; u < 4; ++u) {
        int n = c4 + u;
        Bs[((((n >> 4) * 32) + fragLane(n, k)) << 4) + e] = (__bf16)rb[it][u];
      }
    }
  };

  loadTiles(0);
  for (int k0 = 0; k0 < K; k0 += 32) {
    storeTiles();
    __syncthreads();
    if (k0 + 32 < K) loadTiles(k0 + 32);  // overlap next loads with WMMAs
    v16bf a0 = *(const v16bf*)&As[(((2 * wm + 0) * 32 + lane) << 4)];
    v16bf a1 = *(const v16bf*)&As[(((2 * wm + 1) * 32 + lane) << 4)];
    v16bf b0 = *(const v16bf*)&Bs[(((2 * wn + 0) * 32 + lane) << 4)];
    v16bf b1 = *(const v16bf*)&Bs[(((2 * wn + 1) * 32 + lane) << 4)];
    acc[0][0] = wmma_bf16(a0, b0, acc[0][0]);
    acc[0][1] = wmma_bf16(a0, b1, acc[0][1]);
    acc[1][0] = wmma_bf16(a1, b0, acc[1][0]);
    acc[1][1] = wmma_bf16(a1, b1, acc[1][1]);
    __syncthreads();
  }

#pragma unroll
  for (int i = 0; i < 2; ++i)
#pragma unroll
    for (int j = 0; j < 2; ++j)
#pragma unroll
      for (int r = 0; r < 8; ++r) {
        long m = m0 + 32 * wm + 16 * i + r + (laneHi ? 8 : 0);
        int  n = n0 + 32 * wn + 16 * j + lane15;
        if (interior || n < N) {
          float v = acc[i][j][r];
          if (bias) v += bias[n];
          if (fuse == 1) v = 0.5f * v * (1.f + erff(v * 0.70710678118f));
          Cout[m * N + n] = v;
        }
      }
}

// ---------------------------------------------------------------------------
// Fused SwinV2 window attention: one wave32 per (window, head).
// Cyclic shift handled by address math (read & write the rolled position).
// Cosine attention, learned logit scale, CPB bias, analytic shift mask,
// softmax, P*V. hd = 32, window = 8x8 = 64 tokens.
// ---------------------------------------------------------------------------
__global__ __launch_bounds__(32) void winattn_kernel(
    const float* __restrict__ qkv, const float* __restrict__ logit_scale,
    const float* __restrict__ biasTab, float* __restrict__ outbuf,
    int H, int W, int C, int nh, int shift) {
  const int hd = 32;
  int nWc = W >> 3, nWr = H >> 3;
  int head = blockIdx.x % nh;
  int wid  = blockIdx.x / nh;
  int wc = wid % nWc;
  int wr = (wid / nWc) % nWr;
  int b  = wid / (nWc * nWr);
  int lane = threadIdx.x;
  int laneHi = lane >> 4, lane15 = lane & 15;

  // Fragment-order tiles: [tile][lane][elem]
  __shared__ __bf16 Qs[4 * 32 * 16];        // A-operand, 64x32
  __shared__ __bf16 Ksb[4 * 32 * 16];       // B-operand (K^T), 32x64
  __shared__ __bf16 Vsb[2][2 * 32 * 16];    // B-operand per k-step, 32x32
  __shared__ __bf16 Ps[2][4 * 32 * 16];     // A-operand per k-step, 64x32
  __shared__ float  S[64][64];

  float scale = __expf(fminf(logit_scale[head], 4.60517019f));  // ln(100)
  int C3 = 3 * C;

  // ---- load + cosine-normalize q,k; scatter q,k,v into fragment order ----
  for (int t = lane; t < 64; t += 32) {
    int rr = t >> 3, cc = t & 7;
    int hh = (wr * 8 + rr + shift) % H;
    int ww = (wc * 8 + cc + shift) % W;
    long gtok = ((long)b * H + hh) * W + ww;
    const float* qp = qkv + gtok * C3 + head * hd;
    const float* kp = qp + C;
    const float* vp = qp + 2 * C;
    float qv[32], kv[32], vv[32];
#pragma unroll
    for (int d4 = 0; d4 < 8; ++d4) {
      f32x4 q4 = *(const f32x4*)(qp + 4 * d4);
      f32x4 k4 = *(const f32x4*)(kp + 4 * d4);
      f32x4 v4 = *(const f32x4*)(vp + 4 * d4);
#pragma unroll
      for (int u = 0; u < 4; ++u) {
        qv[4 * d4 + u] = q4[u]; kv[4 * d4 + u] = k4[u]; vv[4 * d4 + u] = v4[u];
      }
    }
    float sq = 0.f, sk = 0.f;
#pragma unroll
    for (int d = 0; d < 32; ++d) { sq += qv[d] * qv[d]; sk += kv[d] * kv[d]; }
    float qi = 1.f / fmaxf(sqrtf(sq), 1e-12f);
    float ki = 1.f / fmaxf(sqrtf(sk), 1e-12f);
    int tt = t >> 4;
#pragma unroll
    for (int khi = 0; khi < 2; ++khi) {
      int blk = tt * 32 + (t & 15) + (khi << 4);
      bf16x8 q0, q1, k0c, k1c;
#pragma unroll
      for (int u = 0; u < 8; ++u) {
        q0[u]  = (__bf16)(qv[8 * khi + u] * qi);
        q1[u]  = (__bf16)(qv[16 + 8 * khi + u] * qi);
        k0c[u] = (__bf16)(kv[8 * khi + u] * ki);
        k1c[u] = (__bf16)(kv[16 + 8 * khi + u] * ki);
      }
      *(bf16x8*)&Qs[(blk << 4) + 0]  = q0;
      *(bf16x8*)&Qs[(blk << 4) + 8]  = q1;
      *(bf16x8*)&Ksb[(blk << 4) + 0] = k0c;
      *(bf16x8*)&Ksb[(blk << 4) + 8] = k1c;
    }
    // V rows scatter: token t is k-index (k' = t&31, ks = t>>5)
    {
      int ks = t >> 5, kp_ = t & 31;
      int e = fragElem(kp_);
      int khi16 = ((kp_ >> 3) & 1) << 4;
#pragma unroll
      for (int n = 0; n < 32; ++n) {
        int blk = (n >> 4) * 32 + (n & 15) + khi16;
        Vsb[ks][(blk << 4) + e] = (__bf16)vv[n];
      }
    }
  }
  __syncthreads();

  const v8f vz = {0.f, 0.f, 0.f, 0.f, 0.f, 0.f, 0.f, 0.f};

  // ---- S = scale * Qn @ Kn^T + bias (+ shift mask): 4x4 WMMA tiles ----
#pragma unroll
  for (int mt = 0; mt < 4; ++mt) {
    v16bf a = *(const v16bf*)&Qs[((mt * 32 + lane) << 4)];
#pragma unroll
    for (int nt = 0; nt < 4; ++nt) {
      v16bf bb = *(const v16bf*)&Ksb[((nt * 32 + lane) << 4)];
      v8f acc = wmma_bf16(a, bb, vz);
#pragma unroll
      for (int r = 0; r < 8; ++r) {
        int m = 16 * mt + r + (laneHi ? 8 : 0);
        int n = 16 * nt + lane15;
        float val = acc[r] * scale + biasTab[head * 4096 + m * 64 + n];
        if (shift > 0) {
          int hm = wr * 8 + (m >> 3), wmc = wc * 8 + (m & 7);
          int hn = wr * 8 + (n >> 3), wnc = wc * 8 + (n & 7);
          int gm = (hm < H - 8 ? 0 : (hm < H - shift ? 1 : 2)) * 3 +
                   (wmc < W - 8 ? 0 : (wmc < W - shift ? 1 : 2));
          int gn = (hn < H - 8 ? 0 : (hn < H - shift ? 1 : 2)) * 3 +
                   (wnc < W - 8 ? 0 : (wnc < W - shift ? 1 : 2));
          if (gm != gn) val -= 100.f;
        }
        S[m][n] = val;
      }
    }
  }
  __syncthreads();

  // ---- row softmax (2 rows/lane) fully in registers; emit P fragments ----
  for (int t = lane; t < 64; t += 32) {
    float rv[64];
#pragma unroll
    for (int j4 = 0; j4 < 16; ++j4) {
      f32x4 v = *(const f32x4*)&S[t][4 * j4];
#pragma unroll
      for (int u = 0; u < 4; ++u) rv[4 * j4 + u] = v[u];
    }
    float mx = -3.0e38f;
#pragma unroll
    for (int j = 0; j < 64; ++j) mx = fmaxf(mx, rv[j]);
    float sum = 0.f;
#pragma unroll
    for (int j = 0; j < 64; ++j) { rv[j] = __expf(rv[j] - mx); sum += rv[j]; }
    float inv = 1.f / sum;
    int tt = t >> 4;
#pragma unroll
    for (int ks = 0; ks < 2; ++ks)
#pragma unroll
      for (int khi = 0; khi < 2; ++khi) {
        int blk = tt * 32 + (t & 15) + (khi << 4);
        bf16x8 c0, c1;
#pragma unroll
        for (int u = 0; u < 8; ++u) {
          c0[u] = (__bf16)(rv[32 * ks + 8 * khi + u] * inv);
          c1[u] = (__bf16)(rv[32 * ks + 16 + 8 * khi + u] * inv);
        }
        *(bf16x8*)&Ps[ks][(blk << 4) + 0] = c0;
        *(bf16x8*)&Ps[ks][(blk << 4) + 8] = c1;
      }
  }
  __syncthreads();

  // ---- O = P @ V : M=64, N=32, K=64 (two K-steps); scatter via unroll map --
#pragma unroll
  for (int mt = 0; mt < 4; ++mt) {
#pragma unroll
    for (int nt = 0; nt < 2; ++nt) {
      v8f acc = vz;
#pragma unroll
      for (int ks = 0; ks < 2; ++ks) {
        v16bf a  = *(const v16bf*)&Ps[ks][((mt * 32 + lane) << 4)];
        v16bf bb = *(const v16bf*)&Vsb[ks][((nt * 32 + lane) << 4)];
        acc = wmma_bf16(a, bb, acc);
      }
#pragma unroll
      for (int r = 0; r < 8; ++r) {
        int m = 16 * mt + r + (laneHi ? 8 : 0);
        int n = 16 * nt + lane15;
        int rr = m >> 3, cc = m & 7;
        int hh = (wr * 8 + rr + shift) % H;
        int ww = (wc * 8 + cc + shift) % W;
        long gtok = ((long)b * H + hh) * W + ww;
        outbuf[gtok * C + head * hd + n] = acc[r];
      }
    }
  }
}

// ---------------------------------------------------------------------------
// LayerNorm over last dim C (one wave per token), optional residual add:
// out = (shortcut ? shortcut : 0) + LN(in) * g + b
// ---------------------------------------------------------------------------
__global__ __launch_bounds__(256) void ln_kernel(
    const float* __restrict__ in, const float* __restrict__ shortcut,
    const float* __restrict__ g, const float* __restrict__ b,
    float* __restrict__ out, int T, int C) {
  int wave = threadIdx.x >> 5, lane = threadIdx.x & 31;
  long t = (long)blockIdx.x * 8 + wave;
  if (t >= T) return;
  const float* row = in + t * C;
  float s = 0.f, sq = 0.f;
  for (int c = lane; c < C; c += 32) { float v = row[c]; s += v; sq += v * v; }
  for (int off = 16; off > 0; off >>= 1) {
    s  += __shfl_xor(s, off, 32);
    sq += __shfl_xor(sq, off, 32);
  }
  float mean = s / (float)C;
  float var  = sq / (float)C - mean * mean;
  float inv  = rsqrtf(var + 1e-5f);
  for (int c = lane; c < C; c += 32) {
    float y = (row[c] - mean) * inv * g[c] + b[c];
    if (shortcut) y += shortcut[t * C + c];
    out[t * C + c] = y;
  }
}

// ---------------------------------------------------------------------------
// Patch embed conv 4x4/s4: NCHW (32,3,256,256) -> NHWC (32,64,64,96) + bias
// ---------------------------------------------------------------------------
__global__ __launch_bounds__(256) void patch_conv_kernel(
    const float* __restrict__ x, const float* __restrict__ kern,
    const float* __restrict__ bias, float* __restrict__ out) {
  long t = (long)blockIdx.x * 256 + threadIdx.x;
  const long total = 32L * 64 * 64 * 96;
  if (t >= total) return;
  int co = (int)(t % 96);
  long pix = t / 96;
  int ow = (int)(pix % 64); pix /= 64;
  int oh = (int)(pix % 64);
  int b  = (int)(pix / 64);
  float s = bias[co];
  for (int kh = 0; kh < 4; ++kh)
    for (int kw = 0; kw < 4; ++kw)
#pragma unroll
      for (int ci = 0; ci < 3; ++ci) {
        float xv = x[(((long)(b * 3 + ci) * 256) + oh * 4 + kh) * 256 + ow * 4 + kw];
        s += xv * kern[((kh * 4 + kw) * 3 + ci) * 96 + co];
      }
  out[t] = s;
}

// ---------------------------------------------------------------------------
// SwinV2 patch-merge 2x2 concat: (B,2H2,2W2,C) -> (B,H2,W2,4C)
// order: [0::2,0::2] [1::2,0::2] [0::2,1::2] [1::2,1::2]
// ---------------------------------------------------------------------------
__global__ __launch_bounds__(256) void merge_concat_kernel(
    const float* __restrict__ x, float* __restrict__ out,
    int B, int H2, int W2, int C) {
  long total = (long)B * H2 * W2 * 4 * C;
  long t = (long)blockIdx.x * 256 + threadIdx.x;
  if (t >= total) return;
  int cc = (int)(t % (4 * C));
  long pix = t / (4 * C);
  int w2 = (int)(pix % W2); pix /= W2;
  int h2 = (int)(pix % H2);
  int b  = (int)(pix / H2);
  int gsel = cc / C, c = cc % C;
  int dh = gsel & 1, dw = gsel >> 1;
  int h = 2 * h2 + dh, w = 2 * w2 + dw;
  out[t] = x[(((long)b * (2 * H2) + h) * (2 * W2) + w) * C + c];
}

// ---------------------------------------------------------------------------
// Continuous position bias: for each (head,i,j) evaluate the 2->512->nh MLP on
// the log-spaced relative coord of (i,j), then 16*sigmoid. biasTab[head][i][j]
// ---------------------------------------------------------------------------
__global__ __launch_bounds__(256) void cpb_kernel(
    const float* __restrict__ w1, const float* __restrict__ b1,
    const float* __restrict__ w2, float* __restrict__ biasTab, int nh) {
  int t = blockIdx.x * 256 + threadIdx.x;
  if (t >= nh * 4096) return;
  int head = t >> 12;
  int ij   = t & 4095;
  int i = ij >> 6, j = ij & 63;
  int dr = (i >> 3) - (j >> 3);
  int dc = (i & 7) - (j & 7);
  float x0 = (float)dr * (8.0f / 7.0f);
  float x1 = (float)dc * (8.0f / 7.0f);
  float t0 = copysignf(log2f(fabsf(x0) + 1.f) * (1.0f / 3.0f), x0);
  float t1 = copysignf(log2f(fabsf(x1) + 1.f) * (1.0f / 3.0f), x1);
  float acc = 0.f;
  for (int h = 0; h < 512; ++h) {
    float hid = fmaxf(t0 * w1[h] + t1 * w1[512 + h] + b1[h], 0.f);
    acc += hid * w2[h * nh + head];
  }
  biasTab[t] = 16.f / (1.f + __expf(-acc));
}

// qkv_b = concat([q_b, zeros, v_b])
__global__ __launch_bounds__(256) void qkvbias_kernel(
    const float* __restrict__ qb, const float* __restrict__ vb,
    float* __restrict__ out, int C) {
  int t = blockIdx.x * 256 + threadIdx.x;
  if (t >= 3 * C) return;
  out[t] = (t < C) ? qb[t] : ((t < 2 * C) ? 0.f : vb[t - 2 * C]);
}

// ---------------------------------------------------------------------------
// Host orchestration
// ---------------------------------------------------------------------------
namespace {
struct Blk {
  const float *qkv_w, *q_b, *v_b, *ls, *cpb_w1, *cpb_b1, *cpb_w2;
  const float *proj_w, *proj_b, *n1g, *n1b, *n2g, *n2b;
  const float *fc1_w, *fc1_b, *fc2_w, *fc2_b;
};
}  // namespace

extern "C" void kernel_launch(void* const* d_in, const int* in_sizes, int n_in,
                              void* d_out, int out_size, void* d_ws, size_t ws_size,
                              hipStream_t stream) {
  (void)out_size; (void)ws_size;
  const float* P[128];
  for (int i = 0; i < n_in && i < 128; ++i) P[i] = (const float*)d_in[i];

  const float *x = nullptr, *patch_k = nullptr, *patch_b = nullptr;
  const float *patch_ng = nullptr, *patch_nb = nullptr;
  const float *ds_red = nullptr, *ds_ng = nullptr, *ds_nb = nullptr;
  const float *norm_g = nullptr, *norm_b = nullptr;
  Blk blk[2][2];

  bool insertion = (n_in > 0 && in_sizes[0] == 32 * 3 * 256 * 256);
  int i = 0;
  if (insertion) {
    // dict insertion order: x; patch{kernel,bias,ng,nb}; stage0{ds:None,
    // blocks}; stage1{ds{red,ng,nb}, blocks}; norm_g; norm_b
    x = P[i++];
    patch_k = P[i++]; patch_b = P[i++]; patch_ng = P[i++]; patch_nb = P[i++];
    auto rd = [&](Blk& k) {
      k.qkv_w = P[i++]; k.q_b = P[i++]; k.v_b = P[i++]; k.ls = P[i++];
      k.cpb_w1 = P[i++]; k.cpb_b1 = P[i++]; k.cpb_w2 = P[i++];
      k.proj_w = P[i++]; k.proj_b = P[i++];
      k.n1g = P[i++]; k.n1b = P[i++]; k.n2g = P[i++]; k.n2b = P[i++];
      k.fc1_w = P[i++]; k.fc1_b = P[i++]; k.fc2_w = P[i++]; k.fc2_b = P[i++];
    };
    rd(blk[0][0]); rd(blk[0][1]);
    ds_red = P[i++]; ds_ng = P[i++]; ds_nb = P[i++];
    rd(blk[1][0]); rd(blk[1][1]);
    norm_g = P[i++]; norm_b = P[i++];
  } else {
    // JAX pytree (sorted keys): norm_b, norm_g, patch{bias,kernel,nb,ng},
    // stages[blocks(sorted leaves), downsample{nb,ng,red}], x
    norm_b = P[i++]; norm_g = P[i++];
    patch_b = P[i++]; patch_k = P[i++]; patch_nb = P[i++]; patch_ng = P[i++];
    auto rd = [&](Blk& k) {
      k.cpb_b1 = P[i++]; k.cpb_w1 = P[i++]; k.cpb_w2 = P[i++];
      k.fc1_b = P[i++]; k.fc1_w = P[i++]; k.fc2_b = P[i++]; k.fc2_w = P[i++];
      k.ls = P[i++];
      k.n1b = P[i++]; k.n1g = P[i++]; k.n2b = P[i++]; k.n2g = P[i++];
      k.proj_b = P[i++]; k.proj_w = P[i++];
      k.q_b = P[i++]; k.qkv_w = P[i++]; k.v_b = P[i++];
    };
    rd(blk[0][0]); rd(blk[0][1]);
    rd(blk[1][0]); rd(blk[1][1]);
    ds_nb = P[i++]; ds_ng = P[i++]; ds_red = P[i++];
    x = P[i++];
  }

  // Workspace layout
  char* ws = (char*)d_ws;
  size_t off = 0;
  auto alloc = [&](size_t bytes) -> float* {
    float* p = (float*)(ws + off);
    off = (off + bytes + 255) & ~((size_t)255);
    return p;
  };
  const long T1 = 131072;  // 32*64*64
  const long T2 = 32768;   // 32*32*32
  float* buf0  = alloc((size_t)T1 * 96 * 4);    // x (current activations)
  float* buf1  = alloc((size_t)T1 * 384 * 4);   // qkv / mlp-hidden / concat
  float* buf2  = alloc((size_t)T1 * 96 * 4);    // attn-out / mlp-out (== T2*384*4)
  float* qkvb  = alloc(768 * 4);                // assembled qkv bias (3C max)
  float* biasT = alloc((size_t)6 * 4096 * 4);   // CPB bias table (nh max 6)

  auto gemm = [&](const float* A, const float* Bw, const float* bias, float* C,
                  long M, int N, int K, int fuse) {
    dim3 g((unsigned)((M + 127) / 128), (unsigned)((N + 63) / 64));
    gemm_bf16_kernel<<<g, 256, 0, stream>>>(A, Bw, bias, C, (int)M, N, K, fuse);
  };
  auto ln = [&](const float* in, const float* sc, const float* g, const float* b,
                float* out, long T, int C) {
    ln_kernel<<<(unsigned)((T + 7) / 8), 256, 0, stream>>>(in, sc, g, b, out, (int)T, C);
  };

  // ---- Patch embed ----
  {
    long total = T1 * 96;
    patch_conv_kernel<<<(unsigned)((total + 255) / 256), 256, 0, stream>>>(
        x, patch_k, patch_b, buf2);
    ln(buf2, nullptr, patch_ng, patch_nb, buf0, T1, 96);
  }

  // ---- Stages ----
  for (int s = 0; s < 2; ++s) {
    if (s == 1) {
      // patch merge: concat -> reduction GEMM (no bias) -> LN
      long total = T2 * 384;
      merge_concat_kernel<<<(unsigned)((total + 255) / 256), 256, 0, stream>>>(
          buf0, buf1, 32, 32, 32, 96);
      gemm(buf1, ds_red, nullptr, buf2, T2, 192, 384, 0);
      ln(buf2, nullptr, ds_ng, ds_nb, buf0, T2, 192);
    }
    long T = s ? T2 : T1;
    int  C = s ? 192 : 96;
    int  H = s ? 32 : 64;
    int  nh = s ? 6 : 3;
    for (int j = 0; j < 2; ++j) {
      const Blk& bk = blk[s][j];
      int shift = j ? 4 : 0;
      // qkv projection
      qkvbias_kernel<<<(unsigned)((3 * C + 255) / 256), 256, 0, stream>>>(
          bk.q_b, bk.v_b, qkvb, C);
      gemm(buf0, bk.qkv_w, qkvb, buf1, T, 3 * C, C, 0);
      // continuous position bias table
      cpb_kernel<<<(unsigned)((nh * 4096 + 255) / 256), 256, 0, stream>>>(
          bk.cpb_w1, bk.cpb_b1, bk.cpb_w2, biasT, nh);
      // fused window attention (one wave per window-head)
      unsigned nwh = (unsigned)((T / 64) * nh);
      winattn_kernel<<<nwh, 32, 0, stream>>>(buf1, bk.ls, biasT, buf2, H, H, C,
                                             nh, shift);
      // output projection + post-norm residual
      gemm(buf2, bk.proj_w, bk.proj_b, buf1, T, C, C, 0);
      ln(buf1, buf0, bk.n1g, bk.n1b, buf0, T, C);
      // MLP (exact GELU) + post-norm residual
      gemm(buf0, bk.fc1_w, bk.fc1_b, buf1, T, 4 * C, C, 1);
      gemm(buf1, bk.fc2_w, bk.fc2_b, buf2, T, C, 4 * C, 0);
      ln(buf2, buf0, bk.n2g, bk.n2b, buf0, T, C);
    }
  }

  // ---- Final norm -> d_out ----
  ln(buf0, nullptr, norm_g, norm_b, (float*)d_out, T2, 192);
}
